// CompiledAutograd0_11879879542251
// MI455X (gfx1250) — matrix operand order repaired and checked by reading, not compiled, not run on hardware
//
#include <hip/hip_runtime.h>
#include <hip/hip_bf16.h>

// Flash-attention forward (causal), fp32 I/O, f16 WMMA compute, CDNA5 wave32.
// B=2 H=16 S=2048 D=64, scale = 0.125.
// Block: 256 threads = 8 waves; each wave owns 32 query rows (2 x 16-row
// sub-tiles sharing K/V fragments). Double-buffered K/V tiles in LDS.
// Wave-uniform guards are forced scalar via readfirstlane -> s_cbranch.

typedef __attribute__((ext_vector_type(16))) _Float16 v16h;
typedef __attribute__((ext_vector_type(8)))  _Float16 v8h;
typedef __attribute__((ext_vector_type(8)))  float    v8f;
typedef __attribute__((ext_vector_type(4)))  float    f32x4;

#define S_LEN   2048
#define D_LEN   64
#define BM      256     // query rows per block
#define BN      32      // kv cols per iteration
#define QP      72      // LDS pitch (halves) for Q rows (64 + 8 pad)
#define KP      72      // LDS pitch for K rows
#define VP      40      // LDS pitch for V^T rows (32 + 8 pad)
#define PP      40      // LDS pitch for P rows

// LDS layout (bytes), region B overlays the one-shot Q staging region A:
//   A: Qs  [256*72 halves]                = 36864
//   B: Ks  [2][32*72 halves]  @ 0         =  9216
//      Vs  [2][64*40 halves]  @ 9216      = 10240
//      Ps  [8][32*40 halves]  @ 19456     = 20480   -> total B = 39936
#define SMEM_BYTES 39936

static __device__ __forceinline__ v16h cat8(v8h lo, v8h hi) {
  return __builtin_shufflevector(lo, hi, 0,1,2,3,4,5,6,7,8,9,10,11,12,13,14,15);
}

// Online softmax update for one 16x32 score tile pair (S0: cols kv0..+15,
// S1: cols kv0+16..+31). Leaves p = exp2(S - mnew) in S0/S1.
static __device__ __forceinline__ void online_softmax(
    v8f& S0, v8f& S1, float mrun[8], float lrun[8],
    v8f Oacc[4], int kv0, int mb, int ln, int hi, int maskTile) {
  const float fscale = 0.125f * 1.44269504088896340736f;  // sm_scale * log2(e)
  float rmax[8];
  if (maskTile) {   // scalar (readfirstlane'd) -> s_cbranch, no exec juggling
    #pragma unroll
    for (int r = 0; r < 8; ++r) {
      int m = mb + r + 8 * hi;
      float s0 = (kv0 + ln      <= m) ? S0[r] * fscale : -3.0e38f;
      float s1 = (kv0 + 16 + ln <= m) ? S1[r] * fscale : -3.0e38f;
      S0[r] = s0; S1[r] = s1;
      rmax[r] = fmaxf(s0, s1);
    }
  } else {
    #pragma unroll
    for (int r = 0; r < 8; ++r) {
      float s0 = S0[r] * fscale;
      float s1 = S1[r] * fscale;
      S0[r] = s0; S1[r] = s1;
      rmax[r] = fmaxf(s0, s1);
    }
  }
  #pragma unroll
  for (int off = 1; off < 16; off <<= 1) {
    #pragma unroll
    for (int r = 0; r < 8; ++r)
      rmax[r] = fmaxf(rmax[r], __shfl_xor(rmax[r], off));
  }
  float alpha[8], rsum[8];
  #pragma unroll
  for (int r = 0; r < 8; ++r) {
    float mn = fmaxf(mrun[r], rmax[r]);
    alpha[r] = exp2f(mrun[r] - mn);
    mrun[r]  = mn;
    float p0 = exp2f(S0[r] - mn);
    float p1 = exp2f(S1[r] - mn);
    S0[r] = p0; S1[r] = p1;
    rsum[r] = p0 + p1;
  }
  #pragma unroll
  for (int off = 1; off < 16; off <<= 1) {
    #pragma unroll
    for (int r = 0; r < 8; ++r)
      rsum[r] += __shfl_xor(rsum[r], off);
  }
  #pragma unroll
  for (int r = 0; r < 8; ++r)
    lrun[r] = lrun[r] * alpha[r] + rsum[r];
  #pragma unroll
  for (int c = 0; c < 4; ++c)
    #pragma unroll
    for (int r = 0; r < 8; ++r)
      Oacc[c][r] *= alpha[r];
}

__launch_bounds__(256, 1)
__global__ void flash_fwd_causal(const float* __restrict__ Q,
                                 const float* __restrict__ K,
                                 const float* __restrict__ V,
                                 float* __restrict__ Out) {
  __shared__ alignas(16) char smem[SMEM_BYTES];
  _Float16* const Qs  = (_Float16*)smem;               // one-shot, overlaid
  _Float16* const KsB = (_Float16*)smem;               // [2][32*KP]
  _Float16* const VsB = (_Float16*)(smem + 9216);      // [2][64*VP]
  _Float16* const PsB = (_Float16*)(smem + 19456);     // [8][32*PP]

  const int tid  = threadIdx.x;
  const int wave = tid >> 5;
  const int lane = tid & 31;
  const int ln   = lane & 15;
  const int hi   = lane >> 4;
  const int q0   = blockIdx.x * BM;
  const int bh   = blockIdx.y;
  const size_t base = (size_t)bh * S_LEN * D_LEN;

  // ---------------- stage Q tile (256x64 f32 -> f16 LDS), then fragments ----
  #pragma unroll
  for (int i = 0; i < 16; ++i) {
    int idx = tid + i * 256;            // float4 index; 16 float4 per row
    int r   = idx >> 4;
    int c4  = (idx & 15) << 2;
    f32x4 f = *(const f32x4*)(Q + base + (size_t)(q0 + r) * D_LEN + c4);
    _Float16* dst = &Qs[r * QP + c4];
    dst[0] = (_Float16)f.x; dst[1] = (_Float16)f.y;
    dst[2] = (_Float16)f.z; dst[3] = (_Float16)f.w;
  }
  __syncthreads();

  // A layout: lane holds row m; halves[0..15]=K[hi*16..hi*16+15]
  v16h aq[2][2];   // [sub 16-row tile][k-step over D]
  #pragma unroll
  for (int sub = 0; sub < 2; ++sub) {
    int mloc = wave * 32 + sub * 16 + ln;
    aq[sub][0] = cat8(*(const v8h*)&Qs[mloc * QP +  0 + hi * 16],
                      *(const v8h*)&Qs[mloc * QP +  8 + hi * 16]);
    aq[sub][1] = cat8(*(const v8h*)&Qs[mloc * QP + 32 + hi * 16],
                      *(const v8h*)&Qs[mloc * QP + 40 + hi * 16]);
  }
  __syncthreads();   // done reading Q region; K/V/P buffers may overwrite it

  // ---------------- softmax / output state -------------------------------
  const int qbase = q0 + wave * 32;                       // wave-uniform
  const int qbase_s = __builtin_amdgcn_readfirstlane(qbase);
  float mrun[2][8], lrun[2][8];
  v8f Oacc[2][4];
  #pragma unroll
  for (int s = 0; s < 2; ++s) {
    #pragma unroll
    for (int r = 0; r < 8; ++r) { mrun[s][r] = -3.0e38f; lrun[s][r] = 0.0f; }
    #pragma unroll
    for (int c = 0; c < 4; ++c)
      Oacc[s][c] = (v8f){0.f,0.f,0.f,0.f,0.f,0.f,0.f,0.f};
  }

  const int kvend = q0 + BM;   // causal: no kv beyond block's last query row
  f32x4 kreg[2], vreg[2];

  // ---- preload K/V tile 0 into buffer 0
  #pragma unroll
  for (int i = 0; i < 2; ++i) {
    int idx = tid + i * 256;
    int r = idx >> 4, c4 = (idx & 15) << 2;
    size_t g = base + (size_t)r * D_LEN + c4;
    kreg[i] = *(const f32x4*)(K + g);
    vreg[i] = *(const f32x4*)(V + g);
  }
  #pragma unroll
  for (int i = 0; i < 2; ++i) {
    int idx = tid + i * 256;
    int r = idx >> 4, c4 = (idx & 15) << 2;
    _Float16* kd = &KsB[r * KP + c4];
    kd[0] = (_Float16)kreg[i].x; kd[1] = (_Float16)kreg[i].y;
    kd[2] = (_Float16)kreg[i].z; kd[3] = (_Float16)kreg[i].w;
    VsB[(c4 + 0) * VP + r] = (_Float16)vreg[i].x;
    VsB[(c4 + 1) * VP + r] = (_Float16)vreg[i].y;
    VsB[(c4 + 2) * VP + r] = (_Float16)vreg[i].z;
    VsB[(c4 + 3) * VP + r] = (_Float16)vreg[i].w;
  }
  __syncthreads();

  for (int kv0 = 0; kv0 < kvend; kv0 += BN) {
    const int  cur  = (kv0 >> 5) & 1;
    const bool more = (kv0 + BN) < kvend;

    // ---- issue next tile's global loads (overlap with WMMA compute)
    if (more) {
      #pragma unroll
      for (int i = 0; i < 2; ++i) {
        int idx = tid + i * 256;
        int r = idx >> 4, c4 = (idx & 15) << 2;
        size_t g = base + (size_t)(kv0 + BN + r) * D_LEN + c4;
        kreg[i] = *(const f32x4*)(K + g);
        vreg[i] = *(const f32x4*)(V + g);
      }
      // prefetch tile t+2 (2 cachelines per 256B row, K and V)
      if (kv0 + 2 * BN < kvend && tid < 128) {
        int arr  = tid >> 6;
        int r    = (tid >> 1) & 31;
        int half = tid & 1;
        const float* p = (arr ? V : K) + base
                       + (size_t)(kv0 + 2 * BN + r) * D_LEN + half * 32;
        __builtin_prefetch(p, 0, 1);
      }
    }

    // ---- compute on current buffer; guards are wave-uniform -> scalar branch
    const int active   = (kv0 <= qbase_s);
    const int maskTile = (kv0 == qbase_s);   // single diagonal tile per wave
    if (active) {
      const _Float16* kb = KsB + cur * (32 * KP);
      const _Float16* vb = VsB + cur * (64 * VP);

      // B fragments of K^T: lane holds column n, 16 contiguous d values
      v16h bk00 = cat8(*(const v8h*)&kb[ln * KP +  0 + hi * 16],
                       *(const v8h*)&kb[ln * KP +  8 + hi * 16]);
      v16h bk01 = cat8(*(const v8h*)&kb[ln * KP + 32 + hi * 16],
                       *(const v8h*)&kb[ln * KP + 40 + hi * 16]);
      v16h bk10 = cat8(*(const v8h*)&kb[(16 + ln) * KP +  0 + hi * 16],
                       *(const v8h*)&kb[(16 + ln) * KP +  8 + hi * 16]);
      v16h bk11 = cat8(*(const v8h*)&kb[(16 + ln) * KP + 32 + hi * 16],
                       *(const v8h*)&kb[(16 + ln) * KP + 40 + hi * 16]);

      _Float16* pw = PsB + wave * (32 * PP);
      #pragma unroll
      for (int sub = 0; sub < 2; ++sub) {
        v8f S0 = (v8f){0.f,0.f,0.f,0.f,0.f,0.f,0.f,0.f};
        v8f S1 = (v8f){0.f,0.f,0.f,0.f,0.f,0.f,0.f,0.f};
        S0 = __builtin_amdgcn_wmma_f32_16x16x32_f16(false, aq[sub][0], false,
                    bk00, (short)0, S0, false, false);
        S0 = __builtin_amdgcn_wmma_f32_16x16x32_f16(false, aq[sub][1], false,
                    bk01, (short)0, S0, false, false);
        S1 = __builtin_amdgcn_wmma_f32_16x16x32_f16(false, aq[sub][0], false,
                    bk10, (short)0, S1, false, false);
        S1 = __builtin_amdgcn_wmma_f32_16x16x32_f16(false, aq[sub][1], false,
                    bk11, (short)0, S1, false, false);

        online_softmax(S0, S1, mrun[sub], lrun[sub], Oacc[sub],
                       kv0, qbase + sub * 16, ln, hi, maskTile);

        // P: C-layout -> LDS (A-layout gathered below)
        #pragma unroll
        for (int r = 0; r < 8; ++r) {
          int m = sub * 16 + r + 8 * hi;
          pw[m * PP + ln]      = (_Float16)S0[r];
          pw[m * PP + 16 + ln] = (_Float16)S1[r];
        }
      }
      // same-wave LDS RAW: DS ops in-order per wave; wait + compiler fence
      asm volatile("s_wait_dscnt 0" ::: "memory");

      v16h ap0 = cat8(*(const v8h*)&pw[ln * PP + hi * 16],
                      *(const v8h*)&pw[ln * PP + 8 + hi * 16]);
      v16h ap1 = cat8(*(const v8h*)&pw[(16 + ln) * PP + hi * 16],
                      *(const v8h*)&pw[(16 + ln) * PP + 8 + hi * 16]);

      #pragma unroll
      for (int c = 0; c < 4; ++c) {
        v16h bv = cat8(*(const v8h*)&vb[(c * 16 + ln) * VP + hi * 16],
                       *(const v8h*)&vb[(c * 16 + ln) * VP + 8 + hi * 16]);
        Oacc[0][c] = __builtin_amdgcn_wmma_f32_16x16x32_f16(false, ap0, false,
                          bv, (short)0, Oacc[0][c], false, false);
        Oacc[1][c] = __builtin_amdgcn_wmma_f32_16x16x32_f16(false, ap1, false,
                          bv, (short)0, Oacc[1][c], false, false);
      }
    }

    // ---- store next tile into alternate buffer, single barrier per iter
    if (more) {
      _Float16* kd0 = KsB + (cur ^ 1) * (32 * KP);
      _Float16* vd0 = VsB + (cur ^ 1) * (64 * VP);
      #pragma unroll
      for (int i = 0; i < 2; ++i) {
        int idx = tid + i * 256;
        int r = idx >> 4, c4 = (idx & 15) << 2;
        _Float16* kd = &kd0[r * KP + c4];
        kd[0] = (_Float16)kreg[i].x; kd[1] = (_Float16)kreg[i].y;
        kd[2] = (_Float16)kreg[i].z; kd[3] = (_Float16)kreg[i].w;
        vd0[(c4 + 0) * VP + r] = (_Float16)vreg[i].x;
        vd0[(c4 + 1) * VP + r] = (_Float16)vreg[i].y;
        vd0[(c4 + 2) * VP + r] = (_Float16)vreg[i].z;
        vd0[(c4 + 3) * VP + r] = (_Float16)vreg[i].w;
      }
    }
    __syncthreads();
  }

  // ---------------- epilogue: normalize + store --------------------------
  #pragma unroll
  for (int sub = 0; sub < 2; ++sub) {
    float linv[8];
    #pragma unroll
    for (int r = 0; r < 8; ++r) linv[r] = 1.0f / lrun[sub][r];
    #pragma unroll
    for (int c = 0; c < 4; ++c) {
      #pragma unroll
      for (int r = 0; r < 8; ++r) {
        int m = qbase + sub * 16 + r + 8 * hi;
        Out[base + (size_t)m * D_LEN + c * 16 + ln] = Oacc[sub][c][r] * linv[r];
      }
    }
  }
}

extern "C" void kernel_launch(void* const* d_in, const int* in_sizes, int n_in,
                              void* d_out, int out_size, void* d_ws, size_t ws_size,
                              hipStream_t stream) {
  (void)in_sizes; (void)n_in; (void)out_size; (void)d_ws; (void)ws_size;
  const float* q = (const float*)d_in[0];
  const float* k = (const float*)d_in[1];
  const float* v = (const float*)d_in[2];
  float* out = (float*)d_out;
  dim3 grid(S_LEN / BM, 2 * 16);   // (8, 32): q-blocks x (B*H)
  flash_fwd_causal<<<grid, 256, 0, stream>>>(q, k, v, out);
}
